// QuantLinearInt8_86268713107932
// MI455X (gfx1250) — 8-bit matrix operand order repaired and checked by reading, compiled, not run on hardware
//
#include <hip/hip_runtime.h>
#include <hip/hip_bf16.h>

typedef __attribute__((ext_vector_type(8))) int   v8i;
typedef __attribute__((ext_vector_type(4))) int   v4i;
typedef __attribute__((ext_vector_type(2))) int   v2i;
typedef __attribute__((ext_vector_type(4))) float v4f;

#define M_DIM 8192
#define K_DIM 4096
#define N_DIM 16384

#define BM 128
#define BN 256
#define BK 64

#if defined(__AMDGCN__) && __has_builtin(__builtin_amdgcn_global_load_async_to_lds_b128)
#define USE_ASYNC 1
#else
#define USE_ASYNC 0
#endif

__global__ __launch_bounds__(256) void quant_rows_kernel(
    const float* __restrict__ x,
    signed char* __restrict__ q,
    float* __restrict__ in_scale) {
  __shared__ float red[256];
  const int row = blockIdx.x;
  const int t   = threadIdx.x;
  const float* xr = x + (size_t)row * K_DIM;

  v4f v[4];
  float amax = 0.0f;
#pragma unroll
  for (int i = 0; i < 4; ++i) {
    v[i] = *(const v4f*)(xr + t * 16 + i * 4);
    amax = fmaxf(amax, fmaxf(fmaxf(fabsf(v[i].x), fabsf(v[i].y)),
                             fmaxf(fabsf(v[i].z), fabsf(v[i].w))));
  }
  red[t] = amax;
  __syncthreads();
#pragma unroll
  for (int s = 128; s > 0; s >>= 1) {
    if (t < s) red[t] = fmaxf(red[t], red[t + s]);
    __syncthreads();
  }
  const float am    = red[0];
  const float scale = am * (1.0f / 127.0f);
  const float inv   = (am > 0.0f) ? (127.0f / am) : 0.0f;
  if (t == 0) in_scale[row] = scale;

  v4i packed;
  int* pk = (int*)&packed;
#pragma unroll
  for (int i = 0; i < 4; ++i) {
    int b0 = (int)rintf(v[i].x * inv);
    int b1 = (int)rintf(v[i].y * inv);
    int b2 = (int)rintf(v[i].z * inv);
    int b3 = (int)rintf(v[i].w * inv);
    b0 = min(127, max(-128, b0));
    b1 = min(127, max(-128, b1));
    b2 = min(127, max(-128, b2));
    b3 = min(127, max(-128, b3));
    pk[i] = (b0 & 0xff) | ((b1 & 0xff) << 8) | ((b2 & 0xff) << 16) |
            ((b3 & 0xff) << 24);
  }
  *(v4i*)(q + (size_t)row * K_DIM + t * 16) = packed;
}

#if USE_ASYNC
typedef __attribute__((address_space(1))) v4i gas_v4i;
typedef __attribute__((address_space(3))) v4i las_v4i;

__device__ __forceinline__ void async_b128(const signed char* g, signed char* l) {
  gas_v4i* gp = (gas_v4i*)g;
  las_v4i* lp = (las_v4i*)l;
  __builtin_amdgcn_global_load_async_to_lds_b128(gp, lp,  0,
                                                  0);
}
__device__ __forceinline__ void wait_async0() {
#if __has_builtin(__builtin_amdgcn_s_wait_asynccnt)
  __builtin_amdgcn_s_wait_asynccnt(0);
#else
  asm volatile("s_wait_asynccnt 0" ::: "memory");
#endif
}
#endif

__device__ __forceinline__ void stage_tiles(
    const signed char* __restrict__ q, const signed char* __restrict__ w,
    signed char* As, signed char* Bs, int m0, int n0, int k0, int tid) {
  const int ra = tid >> 1;
  const int ca = (tid & 1) * 32;
  const signed char* ga = q + (size_t)(m0 + ra) * K_DIM + k0 + ca;
  signed char*       la = As + ra * BK + ca;
  const signed char* gb = w + (size_t)(n0 + tid) * K_DIM + k0;
  signed char*       lb = Bs + tid * BK;
#if USE_ASYNC
  async_b128(ga,      la);
  async_b128(ga + 16, la + 16);
  async_b128(gb,      lb);
  async_b128(gb + 16, lb + 16);
  async_b128(gb + 32, lb + 32);
  async_b128(gb + 48, lb + 48);
#else
  v4i a0 = *(const v4i*)(ga);
  v4i a1 = *(const v4i*)(ga + 16);
  v4i b0 = *(const v4i*)(gb);
  v4i b1 = *(const v4i*)(gb + 16);
  v4i b2 = *(const v4i*)(gb + 32);
  v4i b3 = *(const v4i*)(gb + 48);
  *(v4i*)(la)      = a0;
  *(v4i*)(la + 16) = a1;
  *(v4i*)(lb)      = b0;
  *(v4i*)(lb + 16) = b1;
  *(v4i*)(lb + 32) = b2;
  *(v4i*)(lb + 48) = b3;
#endif
}

__global__ __launch_bounds__(256) void gemm_i8_wmma_kernel(
    const signed char* __restrict__ q,
    const signed char* __restrict__ w,
    const float* __restrict__ in_scale,
    const float* __restrict__ wscale,
    const float* __restrict__ bias,
    __hip_bfloat16* __restrict__ out) {
  __shared__ signed char As[2][BM * BK];
  __shared__ signed char Bs[2][BN * BK];

  const int tid  = threadIdx.x;
  const int wave = tid >> 5;
  const int lane = tid & 31;
  const int half = lane >> 4;
  const int lidx = lane & 15;

  const int waveM = wave >> 2;
  const int waveN = wave & 3;

  const int m0 = blockIdx.y * BM;
  const int n0 = blockIdx.x * BN;

  v8i acc[4][4];
#pragma unroll
  for (int mt = 0; mt < 4; ++mt)
#pragma unroll
    for (int nt = 0; nt < 4; ++nt) acc[mt][nt] = (v8i)(0);

  stage_tiles(q, w, As[0], Bs[0], m0, n0, 0, tid);
#if USE_ASYNC
  wait_async0();
#endif
  __syncthreads();

  int cur = 0;
  for (int k0 = 0; k0 < K_DIM; k0 += BK) {
    const int knext = k0 + BK;
    if (knext < K_DIM)
      stage_tiles(q, w, As[cur ^ 1], Bs[cur ^ 1], m0, n0, knext, tid);

    const signed char* A = As[cur];
    const signed char* B = Bs[cur];

    v8i afrag[4];
#pragma unroll
    for (int mt = 0; mt < 4; ++mt) {
      const signed char* ap = A + (waveM * 64 + mt * 16 + lidx) * BK + half * 8;
      v2i a0 = *(const v2i*)(ap + 0);
      v2i a1 = *(const v2i*)(ap + 16);
      v2i a2 = *(const v2i*)(ap + 32);
      v2i a3 = *(const v2i*)(ap + 48);
      afrag[mt] = (v8i){a0.x, a0.y, a1.x, a1.y, a2.x, a2.y, a3.x, a3.y};
    }

#pragma unroll
    for (int nt = 0; nt < 4; ++nt) {
      const signed char* bp = B + (waveN * 64 + nt * 16 + lidx) * BK;
      v2i b0 = *(const v2i*)(bp + half * 8 + 0);
      v2i b1 = *(const v2i*)(bp + half * 8 + 16);
      v2i b2 = *(const v2i*)(bp + half * 8 + 32);
      v2i b3 = *(const v2i*)(bp + half * 8 + 48);
      v8i bfrag = (v8i){b0.x, b0.y, b1.x, b1.y, b2.x, b2.y, b3.x, b3.y};
#pragma unroll
      for (int mt = 0; mt < 4; ++mt) {
        acc[mt][nt] = __builtin_amdgcn_wmma_i32_16x16x64_iu8(
             true, afrag[mt],
             true, bfrag,
            acc[mt][nt],  false,  false);
      }
    }

#if USE_ASYNC
    if (knext < K_DIM) wait_async0();
#endif
    __syncthreads();
    cur ^= 1;
  }

#pragma unroll
  for (int mt = 0; mt < 4; ++mt) {
    const int mbase = m0 + waveM * 64 + mt * 16 + half * 8;
    v4f s0 = *(const v4f*)(in_scale + mbase);
    v4f s1 = *(const v4f*)(in_scale + mbase + 4);
    const float sc[8] = {s0.x, s0.y, s0.z, s0.w, s1.x, s1.y, s1.z, s1.w};
#pragma unroll
    for (int nt = 0; nt < 4; ++nt) {
      const int n = n0 + waveN * 64 + nt * 16 + lidx;
      const float ws = wscale[n];
      const float bs = bias[n];
#pragma unroll
      for (int i = 0; i < 8; ++i) {
        float v = (float)acc[mt][nt][i] * sc[i] * ws + bs;
        out[(size_t)(mbase + i) * N_DIM + n] = __float2bfloat16(v);
      }
    }
  }
}

extern "C" void kernel_launch(void* const* d_in, const int* in_sizes, int n_in,
                              void* d_out, int out_size, void* d_ws, size_t ws_size,
                              hipStream_t stream) {
  const float*       x      = (const float*)d_in[0];
  const signed char* weight = (const signed char*)d_in[1];
  const float*       wscale = (const float*)d_in[2];
  const float*       bias   = (const float*)d_in[3];
  __hip_bfloat16*    out    = (__hip_bfloat16*)d_out;

  signed char* q        = (signed char*)d_ws;
  float*       in_scale = (float*)((char*)d_ws + (size_t)M_DIM * K_DIM);

  quant_rows_kernel<<<M_DIM, 256, 0, stream>>>(x, q, in_scale);

  dim3 grid(N_DIM / BN, M_DIM / BM);
  gemm_i8_wmma_kernel<<<grid, 256, 0, stream>>>(q, weight, in_scale, wscale,
                                                bias, out);
}
